// own_rnn_67757404062173
// MI455X (gfx1250) — matrix operand order repaired and checked by reading, not compiled, not run on hardware
//
#include <hip/hip_runtime.h>

#define TT 2048
#define BB 256
#define II 64
#define HH 128

typedef __bf16 v16bf __attribute__((ext_vector_type(16)));
typedef __bf16 v8bf  __attribute__((ext_vector_type(8)));
typedef __bf16 v4bf  __attribute__((ext_vector_type(4)));
typedef float  v8f   __attribute__((ext_vector_type(8)));

#define AB_STRIDE 200   // bf16 elems per A row (192 used): 400B row, 16B aligned, bank-friendly
#define HT_STRIDE 20    // fp32 elems per hT column (16 used)

__device__ __forceinline__ float sigmoid_f(float x) {
    return __builtin_amdgcn_rcpf(1.f + __expf(-x));
}
__device__ __forceinline__ float tanh_f(float x) {
    return 2.f * __builtin_amdgcn_rcpf(1.f + __expf(-2.f * x)) - 1.f;
}

// Fill the 6 register-resident B fragments (32x16 KxN each) for one gate.
// B layout used here (contiguous K, same as A): lanes 0-15 hold K=0..15 (VGPR v = K pair 2v,2v+1),
// lanes 16-31 hold K=16..31, N = lane%16.
__device__ __forceinline__ void fill_bfrag(v16bf* f, const float* __restrict__ Wi,
                                           const float* __restrict__ Wh,
                                           int col, int kh) {
#pragma unroll
    for (int kb = 0; kb < 6; ++kb) {
        v16bf v;
#pragma unroll
        for (int j = 0; j < 16; ++j) {
            int K = kb * 32 + kh * 16 + j;                 // combined K row 0..191
            float w = (K < II) ? Wi[K * HH + col] : Wh[(K - II) * HH + col];
            v[j] = (__bf16)w;
        }
        f[kb] = v;
    }
}

__global__ __launch_bounds__(256) void gru_scan_kernel(
    const float* __restrict__ x,
    const float* __restrict__ Wir, const float* __restrict__ Wiz, const float* __restrict__ Win,
    const float* __restrict__ bir, const float* __restrict__ biz, const float* __restrict__ bin,
    const float* __restrict__ Whr, const float* __restrict__ Whz, const float* __restrict__ Whn,
    const float* __restrict__ bhn, const float* __restrict__ Wc,  const float* __restrict__ bc,
    float* __restrict__ out)
{
    // Double-buffered A tile [16 x 192] bf16: cols 0..63 = x_t, cols 64..191 = h_{t-1}
    __shared__ __bf16 Ab[2][16][AB_STRIDE];
    // fp32 h state, transposed [col][row] for contiguous per-lane access
    __shared__ float  hT[2][HH][HT_STRIDE];
    // classifier accumulators (per batch row), double buffered
    __shared__ float  outacc[2][16];

    const int tid  = threadIdx.x;
    const int lane = tid & 31;
    const int w    = tid >> 5;          // wave 0..7
    const int lm   = lane & 15;
    const int kh   = lane >> 4;         // K-half for A/B fragments
    const int r0   = kh * 8;            // C-fragment row base (M=r0..r0+7)
    const int col  = w * 16 + lm;       // hidden column owned by this lane
    const int b0   = blockIdx.x * 16;   // batch tile base

    // ---- register-resident bf16 weights: 18 fragments / wave ----
    v16bf fR[6], fZ[6], fN[6];
    fill_bfrag(fR, Wir, Whr, col, kh);
    fill_bfrag(fZ, Wiz, Whz, col, kh);
    fill_bfrag(fN, Win, Whn, col, kh);

    const float birv = bir[col], bizv = biz[col], binv = bin[col], bhnv = bhn[col];
    const float wcv  = Wc[col];
    const float bcv  = bc[0];

    // ---- prologue: h0 = 0, zero accumulators, stage x_0 ----
    for (int i = tid; i < 2 * 16 * AB_STRIDE; i += 256) ((__bf16*)Ab)[i] = (__bf16)0.f;
    for (int i = tid; i < 2 * HH * HT_STRIDE; i += 256) ((float*)hT)[i] = 0.f;
    if (tid < 32) ((float*)outacc)[tid] = 0.f;
    {
        int row = tid >> 4, c4 = (tid & 15) * 4;
        const float4 xv = *(const float4*)(x + ((size_t)(b0 + row)) * II + c4);
        v4bf cv; cv[0] = (__bf16)xv.x; cv[1] = (__bf16)xv.y;
        cv[2] = (__bf16)xv.z; cv[3] = (__bf16)xv.w;
        *(v4bf*)&Ab[0][row][c4] = cv;
    }

    for (int t = 0; t < TT; ++t) {
        const int cur = t & 1, nxt = cur ^ 1;
        __syncthreads();

        // drain classifier accumulator of step t-1 (then re-zero for step t+1)
        if (w == 0 && lane < 16 && t > 0) {
            float v = outacc[nxt][lane];
            out[(size_t)(t - 1) * BB + b0 + lane] = sigmoid_f(v + bcv);
            outacc[nxt][lane] = 0.f;
        }

        // ---- [16,192] x [192, 3H-slices] via bf16 WMMA, fp32 accum ----
        v8f ar = {}, az = {}, axn = {}, ahn = {};
        const __bf16* arow = &Ab[cur][lm][0];
#pragma unroll
        for (int kb = 0; kb < 6; ++kb) {
            // A fragment (16-bit A 16x32): lane half kh -> K base kb*32 + kh*16,
            // elems 0..15 contiguous
            const __bf16* p = arow + kb * 32 + kh * 16;
            v8bf lo = *(const v8bf*)p;
            v8bf hi = *(const v8bf*)(p + 8);
            v16bf a;
#pragma unroll
            for (int j = 0; j < 8; ++j) { a[j] = lo[j]; a[8 + j] = hi[j]; }
            ar = __builtin_amdgcn_wmma_f32_16x16x32_bf16(false, a, false, fR[kb], (short)0, ar, false, false);
            az = __builtin_amdgcn_wmma_f32_16x16x32_bf16(false, a, false, fZ[kb], (short)0, az, false, false);
            if (kb < 2)  // x @ Win  (K = 0..63)
                axn = __builtin_amdgcn_wmma_f32_16x16x32_bf16(false, a, false, fN[kb], (short)0, axn, false, false);
            else         // h @ Whn  (K = 64..191)
                ahn = __builtin_amdgcn_wmma_f32_16x16x32_bf16(false, a, false, fN[kb], (short)0, ahn, false, false);
        }

        // ---- prefetch x_{t+1} into the other A buffer while WMMAs drain ----
        if (t + 1 < TT) {
            int row = tid >> 4, c4 = (tid & 15) * 4;
            const float4 xv = *(const float4*)(x + ((size_t)(t + 1) * BB + b0 + row) * II + c4);
            v4bf cv; cv[0] = (__bf16)xv.x; cv[1] = (__bf16)xv.y;
            cv[2] = (__bf16)xv.z; cv[3] = (__bf16)xv.w;
            *(v4bf*)&Ab[nxt][row][c4] = cv;
        }

        // ---- gate math (fp32), write h_t (bf16 for next A, fp32 for z*h) ----
        const float* ho = &hT[cur][col][0];
        float*       hw = &hT[nxt][col][0];
        float sv[8];
#pragma unroll
        for (int i = 0; i < 8; ++i) {
            float r = sigmoid_f(ar[i] + birv);
            float z = sigmoid_f(az[i] + bizv);
            float n = tanh_f(axn[i] + binv + r * (ahn[i] + bhnv));
            float h = (1.f - z) * n + z * ho[r0 + i];
            hw[r0 + i] = h;
            Ab[nxt][r0 + i][II + col] = (__bf16)h;
            sv[i] = h * wcv;
        }
        // classifier partial: reduce the 16 columns of this half-wave (wave32 shuffles)
#pragma unroll
        for (int m = 1; m <= 8; m <<= 1) {
#pragma unroll
            for (int i = 0; i < 8; ++i) sv[i] += __shfl_xor(sv[i], m, 32);
        }
        if (lm == 0) {
#pragma unroll
            for (int i = 0; i < 8; ++i) atomicAdd(&outacc[cur][r0 + i], sv[i]);
        }
    }

    __syncthreads();
    if (w == 0 && lane < 16) {
        out[(size_t)(TT - 1) * BB + b0 + lane] =
            sigmoid_f(outacc[(TT - 1) & 1][lane] + bcv);
    }
}

extern "C" void kernel_launch(void* const* d_in, const int* in_sizes, int n_in,
                              void* d_out, int out_size, void* d_ws, size_t ws_size,
                              hipStream_t stream) {
    (void)in_sizes; (void)n_in; (void)out_size; (void)d_ws; (void)ws_size;
    const float* x   = (const float*)d_in[0];
    const float* Wir = (const float*)d_in[1];
    const float* Wiz = (const float*)d_in[2];
    const float* Win = (const float*)d_in[3];
    const float* bir = (const float*)d_in[4];
    const float* biz = (const float*)d_in[5];
    const float* bin = (const float*)d_in[6];
    const float* Whr = (const float*)d_in[7];
    const float* Whz = (const float*)d_in[8];
    const float* Whn = (const float*)d_in[9];
    const float* bhn = (const float*)d_in[10];
    const float* Wc  = (const float*)d_in[11];
    const float* bc  = (const float*)d_in[12];
    float* out = (float*)d_out;

    gru_scan_kernel<<<dim3(BB / 16), dim3(256), 0, stream>>>(
        x, Wir, Wiz, Win, bir, biz, bin, Whr, Whz, Whn, bhn, Wc, bc, out);
}
